// AdaptiveStochasticSNN_35347580846109
// MI455X (gfx1250) — compile-verified, matrix operand order checked by reading, not run on hardware
//
#include <hip/hip_runtime.h>
#include <cstdint>
#include <cstddef>

// ---------------------------------------------------------------------------
// Dims (fixed by the reference)
// ---------------------------------------------------------------------------
#define BB   256      // batch
#define TT   64       // timesteps
#define DD   6400     // input dim
#define HH   1000     // hidden
#define AA   4        // actions
#define MM   (BB*TT)  // GEMM M = 16384
#define KT   (DD/32)  // 200 K-steps

typedef __attribute__((ext_vector_type(16))) __bf16 v16bf;
typedef __attribute__((ext_vector_type(8)))  float  v8f;

union FragCast {
    struct { uint4 lo, hi; } u;   // 32 bytes
    v16bf v;
};

__device__ __forceinline__ v16bf make_frag(uint4 lo, uint4 hi) {
    FragCast f; f.u.lo = lo; f.u.hi = hi; return f.v;
}

// f32 pair -> packed bf16x2: round-half-up bias then one v_perm_b32
__device__ __forceinline__ unsigned bf16pack(float a, float b) {
    unsigned ua = __builtin_bit_cast(unsigned, a) + 0x8000u;
    unsigned ub = __builtin_bit_cast(unsigned, b) + 0x8000u;
    // result = { ub[31:16], ua[31:16] } : bytes {S0.3,S0.2,S1.3,S1.2}
    return __builtin_amdgcn_perm(ub, ua, 0x07060302u);
}

// ---------------------------------------------------------------------------
// Kernel 1: cur1[m, n] = sum_k x[m,k] * W1[n,k] + b1[n]
//   m = b*T + t  (x is [B,T,D] so x flat == row-major [M, K])
//   Workgroup tile: 128(M) x 128(N); 8 waves (4x2), each wave 32(M)x64(N):
//   2 A-frags x 4 B-frags -> 8 v_wmma_f32_16x16x32_bf16 per K-step of 32.
//   Software-pipelined: tile kt+1 is loaded into registers while the WMMAs
//   of tile kt execute; f32->bf16 conversion happens at LDS-store time.
// ---------------------------------------------------------------------------
#define LDSTR 20   // uints per 32-k row: 16 data + 4 pad (80B, keeps 16B align,
                   // stride-20 makes half-wave ds_load_b128 bank-conflict-free)

__global__ __launch_bounds__(256)
void snn_gemm1_wmma(const float* __restrict__ X,
                    const float* __restrict__ W1,
                    const float* __restrict__ b1,
                    float* __restrict__ cur1)
{
    __shared__ __align__(16) unsigned ldsA[128 * LDSTR]; // 10 KB
    __shared__ __align__(16) unsigned ldsB[128 * LDSTR]; // 10 KB

    const int tid     = threadIdx.x;
    const int lane    = tid & 31;
    const int wid     = tid >> 5;
    const int waveM   = (wid & 3) * 32;   // 0,32,64,96
    const int waveN   = (wid >> 2) * 64;  // 0,64
    const int mBase   = blockIdx.y * 128;
    const int nBase   = blockIdx.x * 128;
    const int halfSel = lane >> 4;        // 0: K{0..7,16..23}, 1: K{8..15,24..31}
    const int r       = lane & 15;

    // --- accumulators initialized with bias (each C tile column has fixed n) ---
    int nj[4];
    v8f c0[4], c1[4];
#pragma unroll
    for (int j = 0; j < 4; ++j) {
        nj[j] = nBase + waveN + j * 16 + r;
        const float bias = (nj[j] < HH) ? b1[nj[j]] : 0.0f;
#pragma unroll
        for (int g = 0; g < 8; ++g) { c0[j][g] = bias; c1[j][g] = bias; }
    }

    // --- per-thread staging descriptors (float4 granularity: 8 float4/row) ---
    // Constant element offsets; addresses are re-derived from the kernarg
    // roots every iteration so loads stay in the GLOBAL address space.
    const float4* __restrict__ X4  = (const float4*)X;   // row stride 1600
    const float4* __restrict__ W14 = (const float4*)W1;
    size_t aOff[4], bOff[4];
    bool   bVal[4];
    int    stIdx[4];
#pragma unroll
    for (int i = 0; i < 4; ++i) {
        const int p   = tid + i * 256;   // 0..1023
        const int row = p >> 3;          // 0..127
        const int c4  = p & 7;           // float4 column within 32-k chunk
        aOff[i]  = (size_t)(mBase + row) * (DD / 4) + c4;
        const int nb = nBase + row;
        bVal[i]  = (nb < HH);
        bOff[i]  = (size_t)(bVal[i] ? nb : 0) * (DD / 4) + c4;
        stIdx[i] = row * LDSTR + c4 * 2;
    }

    // --- fragment LDS indices (uint4 units; row = 5 uint4 with padding) ---
    const uint4* A4 = (const uint4*)ldsA;
    const uint4* B4 = (const uint4*)ldsB;
    const int ar0 = (waveM +      r) * 5 + halfSel;
    const int ar1 = (waveM + 16 + r) * 5 + halfSel;
    int br[4];
#pragma unroll
    for (int j = 0; j < 4; ++j) br[j] = (waveN + j * 16 + r) * 5 + halfSel;

    // --- pipelined main loop ---
    float4 va[4], vb[4];
#pragma unroll
    for (int i = 0; i < 4; ++i) {               // prologue: load tile 0
        va[i] = X4[aOff[i]];
        vb[i].x = vb[i].y = vb[i].z = vb[i].w = 0.0f;
        if (bVal[i]) vb[i] = W14[bOff[i]];
    }

    for (int kt = 0; kt < KT; ++kt) {
        // --- pack current tile into LDS ---
#pragma unroll
        for (int i = 0; i < 4; ++i) {
            uint2 pk; pk.x = bf16pack(va[i].x, va[i].y); pk.y = bf16pack(va[i].z, va[i].w);
            *(uint2*)(ldsA + stIdx[i]) = pk;
        }
#pragma unroll
        for (int i = 0; i < 4; ++i) {
            uint2 pk; pk.x = bf16pack(vb[i].x, vb[i].y); pk.y = bf16pack(vb[i].z, vb[i].w);
            *(uint2*)(ldsB + stIdx[i]) = pk;
        }
        __syncthreads();

        // --- issue next tile's global loads (overlap with WMMA below) ---
        if (kt + 1 < KT) {
            const size_t kk = (size_t)(kt + 1) * 8;
#pragma unroll
            for (int i = 0; i < 4; ++i) va[i] = X4[aOff[i] + kk];
#pragma unroll
            for (int i = 0; i < 4; ++i) {
                vb[i].x = vb[i].y = vb[i].z = vb[i].w = 0.0f;
                if (bVal[i]) vb[i] = W14[bOff[i] + kk];
            }
        }

        // --- fragments (ISA 16-bit A-matrix 16x32 lane layout) + 8 WMMA ---
        const v16bf a0 = make_frag(A4[ar0], A4[ar0 + 2]);
        const v16bf a1 = make_frag(A4[ar1], A4[ar1 + 2]);
#pragma unroll
        for (int j = 0; j < 4; ++j) {
            const v16bf bj = make_frag(B4[br[j]], B4[br[j] + 2]);
            c0[j] = __builtin_amdgcn_wmma_f32_16x16x32_bf16(false, a0, false, bj, (short)0, c0[j], false, false);
            c1[j] = __builtin_amdgcn_wmma_f32_16x16x32_bf16(false, a1, false, bj, (short)0, c1[j], false, false);
        }
        __syncthreads();
    }

    // --- store C (VGPR g: M = g + halfSel*8; N = r within 16-tile) ---
#pragma unroll
    for (int j = 0; j < 4; ++j) {
        if (nj[j] < HH) {
#pragma unroll
            for (int g = 0; g < 8; ++g) {
                const int m0r = mBase + waveM + halfSel * 8 + g;
                cur1[(size_t)m0r * HH + nj[j]]        = c0[j][g];
                cur1[(size_t)(m0r + 16) * HH + nj[j]] = c1[j][g];
            }
        }
    }
}

// ---------------------------------------------------------------------------
// Kernel 2: sequential LIF dynamics. One block per batch row b; layer-1 state
// lives in registers (4 neurons/thread), layer-2 (4 units) state on thread 0.
// ---------------------------------------------------------------------------
__global__ __launch_bounds__(256)
void snn_seq_lif(const float* __restrict__ cur1,   // [B*T, H], row m = b*T + t
                 const float* __restrict__ u1,     // [T, B, H]
                 const float* __restrict__ u2,     // [T, B, A]
                 const float* __restrict__ W2,     // [A, H]
                 const float* __restrict__ b2,     // [A]
                 float* __restrict__ out)          // [B, A]
{
    const int b   = blockIdx.x;
    const int tid = threadIdx.x;

    __shared__ float w2s[AA][1024];
    for (int idx = tid; idx < AA * 1024; idx += 256) {
        const int a = idx >> 10, h = idx & 1023;
        w2s[a][h] = (h < HH) ? W2[a * HH + h] : 0.0f;
    }
    __shared__ float red[8][AA];
    __syncthreads();

    float mem[4]  = {0.f, 0.f, 0.f, 0.f};
    float th[4]   = {0.f, 0.f, 0.f, 0.f};
    float m2[AA]  = {0.f, 0.f, 0.f, 0.f};
    float t2[AA]  = {0.f, 0.f, 0.f, 0.f};
    float acc[AA] = {0.f, 0.f, 0.f, 0.f};

    const int lane = tid & 31, wv = tid >> 5;

    for (int t = 0; t < TT; ++t) {
        float part[AA] = {0.f, 0.f, 0.f, 0.f};
#pragma unroll
        for (int j = 0; j < 4; ++j) {
            const int h = tid + j * 256;
            if (h < HH) {
                const float cur = cur1[((size_t)b * TT + t) * HH + h];
                const float u   = u1[((size_t)t * BB + b) * HH + h];
                mem[j] = 0.9f * mem[j] + cur;
                const float z    = mem[j] - (1.0f + th[j]);
                const float prob = 1.0f / (1.0f + __expf(-z));
                const float spk  = (u < prob) ? 1.0f : 0.0f;
                mem[j] *= (1.0f - spk);
                th[j]   = 0.9f * th[j] + 0.05f * spk;
                part[0] = fmaf(spk, w2s[0][h], part[0]);
                part[1] = fmaf(spk, w2s[1][h], part[1]);
                part[2] = fmaf(spk, w2s[2][h], part[2]);
                part[3] = fmaf(spk, w2s[3][h], part[3]);
            }
        }
        // wave32 reduction, then cross-wave via LDS
#pragma unroll
        for (int off = 16; off > 0; off >>= 1) {
#pragma unroll
            for (int a = 0; a < AA; ++a) part[a] += __shfl_down(part[a], off, 32);
        }
        if (lane == 0) {
#pragma unroll
            for (int a = 0; a < AA; ++a) red[wv][a] = part[a];
        }
        __syncthreads();
        if (tid == 0) {
#pragma unroll
            for (int a = 0; a < AA; ++a) {
                float c2 = b2[a];
#pragma unroll
                for (int w = 0; w < 8; ++w) c2 += red[w][a];
                m2[a] = 0.9f * m2[a] + c2;
                const float z    = m2[a] - (1.0f + t2[a]);
                const float prob = 1.0f / (1.0f + __expf(-z));
                const float u    = u2[((size_t)t * BB + b) * AA + a];
                const float spk  = (u < prob) ? 1.0f : 0.0f;
                m2[a] *= (1.0f - spk);
                t2[a]  = 0.9f * t2[a] + 0.05f * spk;
                acc[a] += spk;
            }
        }
        __syncthreads();
    }

    if (tid == 0) {
#pragma unroll
        for (int a = 0; a < AA; ++a) out[b * AA + a] = acc[a] * (1.0f / (float)TT);
    }
}

// ---------------------------------------------------------------------------
// Host launcher
// ---------------------------------------------------------------------------
extern "C" void kernel_launch(void* const* d_in, const int* in_sizes, int n_in,
                              void* d_out, int out_size, void* d_ws, size_t ws_size,
                              hipStream_t stream) {
    const float* x  = (const float*)d_in[0];   // [B,T,D]
    const float* W1 = (const float*)d_in[1];   // [H,D]
    const float* b1 = (const float*)d_in[2];   // [H]
    const float* W2 = (const float*)d_in[3];   // [A,H]
    const float* b2 = (const float*)d_in[4];   // [A]
    const float* u1 = (const float*)d_in[5];   // [T,B,H]
    const float* u2 = (const float*)d_in[6];   // [T,B,A]
    float* out  = (float*)d_out;               // [B,A]
    float* cur1 = (float*)d_ws;                // [B*T, H] f32 = 65.5 MB scratch

    // GEMM: grid.x = N tiles (N-fastest so same-M blocks share x stripe in L2)
    dim3 gGrid(1024 / 128, MM / 128);          // (8, 128)
    snn_gemm1_wmma<<<gGrid, 256, 0, stream>>>(x, W1, b1, cur1);

    // Sequential LIF: one block per batch row
    snn_seq_lif<<<dim3(BB), 256, 0, stream>>>(cur1, u1, u2, W2, b2, out);
}
